// POIEncoder_57140244906591
// MI455X (gfx1250) — matrix operand order repaired and checked by reading, not compiled, not run on hardware
//
#include <hip/hip_runtime.h>
#include <hip/hip_bf16.h>
#include <math.h>

typedef __attribute__((ext_vector_type(16))) _Float16 v16h;
typedef __attribute__((ext_vector_type(8)))  float    v8f;

#define IN_CH 128
#define HC    128
#define HEADS 4
#define CH    32
#define NEG_SLOPE 0.2f
#define LN_EPS 1e-5f

// ---------------------------------------------------------------------------
// 0) init: zero d_out (accumulator), zero denom, set m = -inf
// ---------------------------------------------------------------------------
__global__ void init_kernel(float* __restrict__ out, float* __restrict__ m,
                            float* __restrict__ denom, int n128, int n4) {
  int i = blockIdx.x * blockDim.x + threadIdx.x;
  if (i < n128) out[i] = 0.0f;
  if (i < n4) { m[i] = -__builtin_inff(); denom[i] = 0.0f; }
}

// ---------------------------------------------------------------------------
// 1) xw = x @ W via v_wmma_f32_16x16x32_f16.
//    W (128x128 f32) is staged ONCE per block into LDS as f16, pre-swizzled
//    into B-fragment layout:
//      B frag for (kb32, tile t): lane holds col = t*16+(lane&15),
//      K = kb32*32 + 16*(lane>>4) + j  (j = 0..15, contiguous in LDS)
//    so each lane's 16 f16 are one 32-byte contiguous LDS read.
//    A frag (16x32 f16): lane<16: row=lane,  K in {kb..kb+15}
//                        lane>=16: row=lane-16, K in {kb+16..kb+31}
//    C/D (16x16 f32, 8 VGPRs): row = tileM + j + 8*(lane>>4), col = tileN+(lane&15)
// ---------------------------------------------------------------------------
__global__ __launch_bounds__(256) void gemm_xw_kernel(
    const float* __restrict__ x, const float* __restrict__ W,
    float* __restrict__ xw, int nrows) {
  __shared__ _Float16 Wl[4][8][32][16];   // [kb32][tile][lane][j] : 32 KB

  // --- cooperative stage: W[k][c] -> Wl[k>>5][c>>4][((k>>4)&1)*16+(c&15)][k&15]
  for (int idx = threadIdx.x; idx < (IN_CH * HC) / 4; idx += 256) {
    const int k  = idx >> 5;            // row of W
    const int c0 = (idx & 31) * 4;      // 4 consecutive cols
    const float4 w4 = *(const float4*)(W + (size_t)k * HC + c0);
    const int kb32 = k >> 5, half = (k >> 4) & 1, j = k & 15;
    const float wv[4] = {w4.x, w4.y, w4.z, w4.w};
#pragma unroll
    for (int q = 0; q < 4; ++q) {
      const int c = c0 + q;
      Wl[kb32][c >> 4][half * 16 + (c & 15)][j] = (_Float16)wv[q];
    }
  }
  __syncthreads();

  const int lane  = (int)(threadIdx.x & 31);
  const int tileM = (int)(blockIdx.x * 8 + (threadIdx.x >> 5)) * 16;
  if (tileM >= nrows) return;           // wave-uniform exit (after barrier)

  const int half = lane >> 4;
  const int l15  = lane & 15;
  int arow = tileM + l15;
  if (arow >= nrows) arow = nrows - 1;  // safe clamp (N%16==0 in practice)
  const float* __restrict__ xrow = x + (size_t)arow * IN_CH;

  v8f acc[8];
#pragma unroll
  for (int t = 0; t < 8; ++t) acc[t] = (v8f){};

#pragma unroll
  for (int kb32 = 0; kb32 < 4; ++kb32) {
    const int kb = kb32 * 32;
    // --- A fragment: 4 aligned float4 loads + pack to f16 ---
    const float4 a0 = *(const float4*)(xrow + kb + half * 16);
    const float4 a1 = *(const float4*)(xrow + kb + half * 16 + 4);
    const float4 a2 = *(const float4*)(xrow + kb + half * 16 + 8);
    const float4 a3 = *(const float4*)(xrow + kb + half * 16 + 12);
    v16h afrag;
    afrag[0]  = (_Float16)a0.x; afrag[1]  = (_Float16)a0.y;
    afrag[2]  = (_Float16)a0.z; afrag[3]  = (_Float16)a0.w;
    afrag[4]  = (_Float16)a1.x; afrag[5]  = (_Float16)a1.y;
    afrag[6]  = (_Float16)a1.z; afrag[7]  = (_Float16)a1.w;
    afrag[8]  = (_Float16)a2.x; afrag[9]  = (_Float16)a2.y;
    afrag[10] = (_Float16)a2.z; afrag[11] = (_Float16)a2.w;
    afrag[12] = (_Float16)a3.x; afrag[13] = (_Float16)a3.y;
    afrag[14] = (_Float16)a3.z; afrag[15] = (_Float16)a3.w;

#pragma unroll
    for (int t = 0; t < 8; ++t) {
      const v16h bfrag = *(const v16h*)&Wl[kb32][t][lane][0];  // 32B LDS read
      acc[t] = __builtin_amdgcn_wmma_f32_16x16x32_f16(
          /*neg_a=*/false, afrag, /*neg_b=*/false, bfrag,
          /*c_mod=*/(short)0, acc[t], /*reuse_a=*/false, /*reuse_b=*/false);
    }
  }

  // --- store D tiles (coalesced across lanes within each 16-col tile) ---
#pragma unroll
  for (int t = 0; t < 8; ++t) {
#pragma unroll
    for (int j = 0; j < 8; ++j) {
      const int r = tileM + j + 8 * half;
      if (r < nrows) xw[(size_t)r * HC + t * 16 + l15] = acc[t][j];
    }
  }
}

// ---------------------------------------------------------------------------
// 2) per-node attention scores: a_src[n,h] = <xw[n,h,:], att_src[h,:]>, same dst
// ---------------------------------------------------------------------------
__global__ void scores_kernel(const float* __restrict__ xw,
                              const float* __restrict__ att_src,
                              const float* __restrict__ att_dst,
                              float* __restrict__ a_src, float* __restrict__ a_dst,
                              int nh) {
  int i = blockIdx.x * blockDim.x + threadIdx.x;
  if (i >= nh) return;
  const int h = i & 3;
  const float4* p  = (const float4*)(xw + (size_t)(i >> 2) * HC + h * CH);
  const float4* sa = (const float4*)(att_src + h * CH);
  const float4* da = (const float4*)(att_dst + h * CH);
  float ss = 0.0f, sd = 0.0f;
#pragma unroll
  for (int q = 0; q < CH / 4; ++q) {
    const float4 v = p[q], a = sa[q], b = da[q];
    ss += v.x * a.x + v.y * a.y + v.z * a.z + v.w * a.w;
    sd += v.x * b.x + v.y * b.y + v.z * b.z + v.w * b.w;
  }
  a_src[i] = ss;
  a_dst[i] = sd;
}

// ---------------------------------------------------------------------------
// float atomic max via int punning (m initialized to -inf)
// ---------------------------------------------------------------------------
__device__ __forceinline__ void atomicMaxFloat(float* addr, float val) {
  if (val >= 0.0f) atomicMax((int*)addr, __float_as_int(val));
  else             atomicMin((unsigned int*)addr, (unsigned int)__float_as_int(val));
}

__device__ __forceinline__ void load_edge(const long long* __restrict__ ei,
                                          int e, int E_, int& s, int& d) {
  if (e < E_) { s = (int)ei[e]; d = (int)ei[(size_t)E_ + e]; }
  else        { s = e - E_; d = s; }          // self loops appended
}

// ---------------------------------------------------------------------------
// 3) segment-max of leaky_relu logits, grouped by destination
// ---------------------------------------------------------------------------
__global__ void edge_max_kernel(const long long* __restrict__ ei,
                                const float* __restrict__ a_src,
                                const float* __restrict__ a_dst,
                                float* __restrict__ m, int E_, int total) {
  int e = blockIdx.x * blockDim.x + threadIdx.x;
  if (e >= total) return;
  int s, d; load_edge(ei, e, E_, s, d);
  const float4 as = *(const float4*)(a_src + 4 * (size_t)s);
  const float4 ad = *(const float4*)(a_dst + 4 * (size_t)d);
  float v[4] = {as.x + ad.x, as.y + ad.y, as.z + ad.z, as.w + ad.w};
#pragma unroll
  for (int h = 0; h < HEADS; ++h) {
    float lv = v[h] > 0.0f ? v[h] : NEG_SLOPE * v[h];
    atomicMaxFloat(&m[4 * (size_t)d + h], lv);
  }
}

// ---------------------------------------------------------------------------
// 4) segment-sum of exp(logit - max)
// ---------------------------------------------------------------------------
__global__ void edge_sum_kernel(const long long* __restrict__ ei,
                                const float* __restrict__ a_src,
                                const float* __restrict__ a_dst,
                                const float* __restrict__ m,
                                float* __restrict__ denom, int E_, int total) {
  int e = blockIdx.x * blockDim.x + threadIdx.x;
  if (e >= total) return;
  int s, d; load_edge(ei, e, E_, s, d);
  const float4 as = *(const float4*)(a_src + 4 * (size_t)s);
  const float4 ad = *(const float4*)(a_dst + 4 * (size_t)d);
  const float4 mm = *(const float4*)(m + 4 * (size_t)d);
  float v[4]  = {as.x + ad.x, as.y + ad.y, as.z + ad.z, as.w + ad.w};
  float mx[4] = {mm.x, mm.y, mm.z, mm.w};
#pragma unroll
  for (int h = 0; h < HEADS; ++h) {
    float lv = v[h] > 0.0f ? v[h] : NEG_SLOPE * v[h];
    atomicAdd(&denom[4 * (size_t)d + h], __expf(lv - mx[h]));
  }
}

// ---------------------------------------------------------------------------
// 5) weighted scatter: one wave per edge, lane -> channel within each head.
//    out[dst, h*32+lane] += alpha[h] * xw[src, h*32+lane]
// ---------------------------------------------------------------------------
__global__ __launch_bounds__(256) void edge_scatter_kernel(
    const long long* __restrict__ ei,
    const float* __restrict__ a_src, const float* __restrict__ a_dst,
    const float* __restrict__ m, const float* __restrict__ denom,
    const float* __restrict__ xw, float* __restrict__ out, int E_, int total) {
  const int wid  = (int)((blockIdx.x * blockDim.x + threadIdx.x) >> 5);
  const int lane = (int)(threadIdx.x & 31);
  if (wid >= total) return;
  int s, d; load_edge(ei, wid, E_, s, d);
  const float* xs = xw + (size_t)s * HC;
  float* od = out + (size_t)d * HC;
#pragma unroll
  for (int h = 0; h < HEADS; ++h) {
    float v = a_src[4 * (size_t)s + h] + a_dst[4 * (size_t)d + h];
    v = v > 0.0f ? v : NEG_SLOPE * v;
    float alpha = __expf(v - m[4 * (size_t)d + h]) /
                  (denom[4 * (size_t)d + h] + 1e-16f);
    int c = h * CH + lane;
    atomicAdd(&od[c], alpha * xs[c]);
  }
}

// ---------------------------------------------------------------------------
// 6) bias + LayerNorm, in place on d_out. One wave32 per node, 4 ch per lane.
// ---------------------------------------------------------------------------
__global__ __launch_bounds__(256) void ln_kernel(
    float* __restrict__ out, const float* __restrict__ bias,
    const float* __restrict__ gamma, const float* __restrict__ beta, int n) {
  const int wid  = (int)((blockIdx.x * blockDim.x + threadIdx.x) >> 5);
  const int lane = (int)(threadIdx.x & 31);
  if (wid >= n) return;
  float v[HEADS];
  float sum = 0.0f;
#pragma unroll
  for (int h = 0; h < HEADS; ++h) {
    int c = h * CH + lane;
    v[h] = out[(size_t)wid * HC + c] + bias[c];
    sum += v[h];
  }
#pragma unroll
  for (int off = 16; off > 0; off >>= 1) sum += __shfl_xor(sum, off, 32);
  const float mu = sum * (1.0f / (float)HC);
  float sq = 0.0f;
#pragma unroll
  for (int h = 0; h < HEADS; ++h) { float dv = v[h] - mu; sq += dv * dv; }
#pragma unroll
  for (int off = 16; off > 0; off >>= 1) sq += __shfl_xor(sq, off, 32);
  const float inv = rsqrtf(sq * (1.0f / (float)HC) + LN_EPS);
#pragma unroll
  for (int h = 0; h < HEADS; ++h) {
    int c = h * CH + lane;
    out[(size_t)wid * HC + c] = gamma[c] * (v[h] - mu) * inv + beta[c];
  }
}

// ---------------------------------------------------------------------------
// launcher
// ---------------------------------------------------------------------------
extern "C" void kernel_launch(void* const* d_in, const int* in_sizes, int n_in,
                              void* d_out, int out_size, void* d_ws, size_t ws_size,
                              hipStream_t stream) {
  const float*     x        = (const float*)d_in[0];
  const long long* ei       = (const long long*)d_in[1];  // int64 [2, E]
  // d_in[2] = edge_weight : unused by GATConv reference
  const float*     W        = (const float*)d_in[3];
  const float*     att_src  = (const float*)d_in[4];
  const float*     att_dst  = (const float*)d_in[5];
  const float*     bias     = (const float*)d_in[6];
  const float*     gamma    = (const float*)d_in[7];
  const float*     beta     = (const float*)d_in[8];
  float*           out      = (float*)d_out;

  const int N  = in_sizes[0] / IN_CH;
  const int E  = in_sizes[1] / 2;
  const int total = E + N;             // edges + self loops

  // workspace layout (floats)
  float* xw    = (float*)d_ws;                 // N*128
  float* a_src = xw    + (size_t)N * HC;       // N*4
  float* a_dst = a_src + (size_t)N * 4;        // N*4
  float* m     = a_dst + (size_t)N * 4;        // N*4
  float* denom = m     + (size_t)N * 4;        // N*4

  const int B = 256;

  // 0) init accumulators
  {
    int n128 = N * HC, n4 = N * 4;
    int g = (n128 + B - 1) / B;
    init_kernel<<<g, B, 0, stream>>>(out, m, denom, n128, n4);
  }
  // 1) GEMM (WMMA, W staged in LDS)
  {
    int nwaves = (N + 15) / 16;
    int g = (nwaves + 7) / 8;               // 8 waves / 256-thread block
    gemm_xw_kernel<<<g, B, 0, stream>>>(x, W, xw, N);
  }
  // 2) scores
  {
    int nh = N * 4;
    int g = (nh + B - 1) / B;
    scores_kernel<<<g, B, 0, stream>>>(xw, att_src, att_dst, a_src, a_dst, nh);
  }
  // 3) segment max
  {
    int g = (total + B - 1) / B;
    edge_max_kernel<<<g, B, 0, stream>>>(ei, a_src, a_dst, m, E, total);
  }
  // 4) segment sum
  {
    int g = (total + B - 1) / B;
    edge_sum_kernel<<<g, B, 0, stream>>>(ei, a_src, a_dst, m, denom, E, total);
  }
  // 5) weighted scatter (wave per edge)
  {
    long long threads = (long long)total * 32;
    int g = (int)((threads + B - 1) / B);
    edge_scatter_kernel<<<g, B, 0, stream>>>(ei, a_src, a_dst, m, denom, xw, out, E, total);
  }
  // 6) bias + LayerNorm
  {
    long long threads = (long long)N * 32;
    int g = (int)((threads + B - 1) / B);
    ln_kernel<<<g, B, 0, stream>>>(out, bias, gamma, beta, N);
  }
}
